// S6Layer_40183714022013
// MI455X (gfx1250) — compile-verified, matrix operand order checked by reading, not run on hardware
//
#include <hip/hip_runtime.h>
#include <hip/hip_bf16.h>

// ---------------------------------------------------------------------------
// S6 (Mamba) layer forward on gfx1250.
//   D_MODEL=1024, D_STATE=16, D_CONV=4, D_INNER=2048, DT_RANK=64, B=2, L=4096
//
// GEMMs use bf16x3 split-precision f32 emulation on the bf16 matrix pipe:
//   A*B ~= Ahi*Bhi + Ahi*Blo + Alo*Bhi   (f32 accumulate)
// Operands are pre-split ONCE into packed bf16 hi/lo planes in HBM
// (B operands also pre-transposed to [N][K] and zero-padded to the tile),
// so GEMM staging is a pure copy: global -> LDS via async-DMA
// (GLOBAL_LOAD_ASYNC_TO_LDS_B128, ASYNCcnt) with double-buffered ping-pong.
// ---------------------------------------------------------------------------

typedef __bf16 bf16;
typedef __attribute__((ext_vector_type(8)))  bf16  v8bf;
typedef __attribute__((ext_vector_type(16))) bf16  v16bf;
typedef __attribute__((ext_vector_type(8)))  float v8f;
typedef __attribute__((ext_vector_type(4)))  int   v4i;

#define BM  64
#define BN  128
#define BKC 32          // K per LDS stage = one WMMA K-chunk
#define LROW 40         // 32 + 8 bf16 pad: 80B stride (16B aligned, 20-bank step)

#if __has_builtin(__builtin_amdgcn_global_load_async_to_lds_b128)
#define USE_ASYNC_LDS 1
#else
#define USE_ASYNC_LDS 0
#endif

__device__ __forceinline__ void ld_lds_b128(const bf16* g, bf16* l) {
#if USE_ASYNC_LDS
    __builtin_amdgcn_global_load_async_to_lds_b128((v4i*)g, (v4i*)l, 0, 0);
#else
    *(v8bf*)l = *(const v8bf*)g;
#endif
}

__device__ __forceinline__ void wait_async_tail() {   // next stage (6) may remain
#if USE_ASYNC_LDS
#if __has_builtin(__builtin_amdgcn_s_wait_asynccnt)
    __builtin_amdgcn_s_wait_asynccnt(6);
#else
    asm volatile("s_wait_asynccnt 6" ::: "memory");
#endif
#endif
}
__device__ __forceinline__ void wait_async_all() {
#if USE_ASYNC_LDS
#if __has_builtin(__builtin_amdgcn_s_wait_asynccnt)
    __builtin_amdgcn_s_wait_asynccnt(0);
#else
    asm volatile("s_wait_asynccnt 0" ::: "memory");
#endif
#endif
}

__device__ __forceinline__ v16bf cat8(v8bf a, v8bf b) {
    return __builtin_shufflevector(a, b, 0,1,2,3,4,5,6,7,8,9,10,11,12,13,14,15);
}

// C[M,N] = f(A[M,K] @ B[K,N]); A given as hi/lo bf16 planes [M x lda],
// B given as transposed hi/lo planes Bt[Npad x ldbt] (zero-padded rows).
// Requires M%64==0, K%32==0. epilogue: 0=store, 1=softplus(v + bias[col]).
__global__ __launch_bounds__(256) void gemm_bf16x3_wmma(
    const bf16* __restrict__ Ahg, const bf16* __restrict__ Alg, long lda,
    const bf16* __restrict__ Bhg, const bf16* __restrict__ Blg, long ldbt,
    float* __restrict__ C, long ldc,
    int M, int N, int K,
    const float* __restrict__ bias, int epilogue)
{
    __shared__ bf16 Ah[2][BM][LROW];   // 10240 B
    __shared__ bf16 Al[2][BM][LROW];   // 10240 B
    __shared__ bf16 Bh[2][BN][LROW];   // 20480 B   (Bt rows: [n][k])
    __shared__ bf16 Bl[2][BN][LROW];   // 20480 B   -> 61440 B total

    const int tid  = threadIdx.x;
    const int lane = tid & 31;
    const int wave = tid >> 5;         // 0..7
    const int wm   = wave >> 2;        // 0..1
    const int wn   = wave & 3;         // 0..3
    const int l16  = lane & 15;
    const int half = lane >> 4;        // K sub-block select (16-bit frag layout)
    const long rowBlock = (long)blockIdx.y * BM;
    const long colBlock = (long)blockIdx.x * BN;

    // per-thread staging coordinates (b128 = 8 bf16 chunks)
    const int ar  = tid >> 2;          // 0..63   A row
    const int ac  = (tid & 3) << 3;    // 0,8,16,24  A col (bf16)
    const int br0 = tid >> 2;          // 0..63   Bt rows (two batches)
    const int br1 = 64 + (tid >> 2);   // 64..127
    const int bc  = (tid & 3) << 3;

    auto stage = [&](int buf, int k0) {
        ld_lds_b128(&Ahg[(rowBlock + ar) * lda + k0 + ac], &Ah[buf][ar][ac]);
        ld_lds_b128(&Alg[(rowBlock + ar) * lda + k0 + ac], &Al[buf][ar][ac]);
        ld_lds_b128(&Bhg[(colBlock + br0) * ldbt + k0 + bc], &Bh[buf][br0][bc]);
        ld_lds_b128(&Blg[(colBlock + br0) * ldbt + k0 + bc], &Bl[buf][br0][bc]);
        ld_lds_b128(&Bhg[(colBlock + br1) * ldbt + k0 + bc], &Bh[buf][br1][bc]);
        ld_lds_b128(&Blg[(colBlock + br1) * ldbt + k0 + bc], &Bl[buf][br1][bc]);
    };

    v8f acc[2][2];
    #pragma unroll
    for (int i = 0; i < 2; ++i)
        #pragma unroll
        for (int j = 0; j < 2; ++j)
            acc[i][j] = (v8f)(0.0f);

    const int S = K / BKC;
    stage(0, 0);

    for (int s = 0; s < S; ++s) {
        const int buf = s & 1;
        if (s + 1 < S) {               // prefetch next slab into the other buffer
            stage(buf ^ 1, (s + 1) * BKC);
            wait_async_tail();         // in-order: current slab (first 6) landed
        } else {
            wait_async_all();
        }
        __syncthreads();

        const int off = half * 8;      // bf16 units; frag = [off,+8) ++ [off+16,+8)
        v16bf ah[2], al[2], bh[2], bl[2];
        #pragma unroll
        for (int tm = 0; tm < 2; ++tm) {
            int row = wm * 32 + tm * 16 + l16;
            ah[tm] = cat8(*(const v8bf*)&Ah[buf][row][off],
                          *(const v8bf*)&Ah[buf][row][off + 16]);
            al[tm] = cat8(*(const v8bf*)&Al[buf][row][off],
                          *(const v8bf*)&Al[buf][row][off + 16]);
        }
        #pragma unroll
        for (int tn = 0; tn < 2; ++tn) {
            int col = wn * 32 + tn * 16 + l16;
            bh[tn] = cat8(*(const v8bf*)&Bh[buf][col][off],
                          *(const v8bf*)&Bh[buf][col][off + 16]);
            bl[tn] = cat8(*(const v8bf*)&Bl[buf][col][off],
                          *(const v8bf*)&Bl[buf][col][off + 16]);
        }
        #pragma unroll
        for (int tm = 0; tm < 2; ++tm) {
            #pragma unroll
            for (int tn = 0; tn < 2; ++tn) {
                acc[tm][tn] = __builtin_amdgcn_wmma_f32_16x16x32_bf16(
                    false, ah[tm], false, bh[tn], (short)0, acc[tm][tn], false, false);
                acc[tm][tn] = __builtin_amdgcn_wmma_f32_16x16x32_bf16(
                    false, ah[tm], false, bl[tn], (short)0, acc[tm][tn], false, false);
                acc[tm][tn] = __builtin_amdgcn_wmma_f32_16x16x32_bf16(
                    false, al[tm], false, bh[tn], (short)0, acc[tm][tn], false, false);
            }
        }
        __syncthreads();
    }

    // ---- epilogue: VGPR j of C/D tile holds row (j + half*8), col = l16 ----
    #pragma unroll
    for (int tm = 0; tm < 2; ++tm) {
        #pragma unroll
        for (int tn = 0; tn < 2; ++tn) {
            long col = colBlock + wn * 32 + tn * 16 + l16;
            if (col < N) {
                long rbase = rowBlock + wm * 32 + tm * 16 + half * 8;
                float bv = (epilogue == 1) ? bias[col] : 0.0f;
                #pragma unroll
                for (int j = 0; j < 8; ++j) {
                    float v = acc[tm][tn][j];
                    if (epilogue == 1) {
                        v += bv;
                        v = (v > 20.0f) ? v : log1pf(__expf(v));   // softplus
                    }
                    C[(rbase + j) * ldc + col] = v;
                }
            }
        }
    }
}

// ---------------------------------------------------------------------------
// f32 -> bf16 hi/lo split (packed planes), elementwise.
__global__ __launch_bounds__(256) void split_kernel(
    const float* __restrict__ src, bf16* __restrict__ hi, bf16* __restrict__ lo,
    long n)
{
    long i = (long)blockIdx.x * blockDim.x + threadIdx.x;
    if (i < n) {
        float v = src[i];
        bf16 h = (bf16)v;
        hi[i] = h;
        lo[i] = (bf16)(v - (float)h);
    }
}

// f32 [K x N] -> transposed bf16 hi/lo planes [Npad x K], rows >= N zero-filled.
__global__ __launch_bounds__(256) void splitT_kernel(
    const float* __restrict__ src, bf16* __restrict__ hi, bf16* __restrict__ lo,
    int K, int N, long total /* = Npad*K */)
{
    long i = (long)blockIdx.x * blockDim.x + threadIdx.x;
    if (i < total) {
        int k = (int)(i % K);
        int n = (int)(i / K);
        float v = (n < N) ? src[(long)k * N + n] : 0.0f;
        bf16 h = (bf16)v;
        hi[i] = h;
        lo[i] = (bf16)(v - (float)h);
    }
}

// ---------------------------------------------------------------------------
// Depthwise causal conv (taps=4) + silu.
// x_proj lives in xz[:, 0:2048] (row stride 4096). Output x_conv [row*2048+d].
__global__ __launch_bounds__(256) void conv_silu_kernel(
    const float* __restrict__ xz,
    const float* __restrict__ conv_w,   // [2048][1][4]
    const float* __restrict__ conv_b,   // [2048]
    float* __restrict__ x_conv)
{
    const long idx = (long)blockIdx.x * blockDim.x + threadIdx.x;
    const int  d   = (int)(idx & 2047);
    const long row = idx >> 11;          // b*L + t
    const int  t   = (int)(row & 4095);

    float acc = conv_b[d];
    #pragma unroll
    for (int j = 0; j < 4; ++j) {
        int tt = t - 3 + j;
        if (tt >= 0)
            acc = fmaf(xz[(row - 3 + j) * 4096 + d], conv_w[d * 4 + j], acc);
    }
    float sig = 1.0f / (1.0f + __expf(-acc));
    x_conv[idx] = acc * sig;             // silu
}

// ---------------------------------------------------------------------------
// Selective scan. One thread per (b, d) channel; h[16] in registers.
// dt lives in xz[:, 0:2048], z in xz[:, 2048:4096] (row stride 4096).
// B_t / C_t come from x_dbl[row*96 + 64/80 + n] (wave-uniform -> SMEM loads).
// y (gated by silu(z)) overwrites x_conv in place.
__global__ __launch_bounds__(256) void s6_scan_kernel(
    const float* __restrict__ xz,
    float* __restrict__ x_conv,
    const float* __restrict__ x_dbl,
    const float* __restrict__ A_log,
    const float* __restrict__ Dvec,
    const float* __restrict__ h0,
    float* __restrict__ fs_out)
{
    const int d = blockIdx.x * blockDim.x + threadIdx.x;   // 0..2047
    const int b = blockIdx.y;                              // 0..1
    const int L = 4096;

    float Arow[16], h[16];
    #pragma unroll
    for (int n = 0; n < 16; ++n) {
        Arow[n] = -__expf(A_log[d * 16 + n]);
        h[n] = h0[((long)b * 2048 + d) * 16 + n];
    }
    const float Dd = Dvec[d];

    for (int t = 0; t < L; ++t) {
        const long r = (long)b * L + t;
        const float dtv = xz[r * 4096 + d];
        const float xv  = x_conv[r * 2048 + d];
        const float zv  = xz[r * 4096 + 2048 + d];
        const float dtx = dtv * xv;
        const float* bc = x_dbl + r * 96;   // wave-uniform -> scalar loads
        float y = Dd * xv;
        #pragma unroll
        for (int n = 0; n < 16; ++n) {
            float Bn = bc[64 + n];
            float Cn = bc[80 + n];
            h[n] = __expf(dtv * Arow[n]) * h[n] + dtx * Bn;
            y = fmaf(h[n], Cn, y);
        }
        float sig = 1.0f / (1.0f + __expf(-zv));
        x_conv[r * 2048 + d] = y * (zv * sig);   // y * silu(z)
    }
    #pragma unroll
    for (int n = 0; n < 16; ++n)
        fs_out[((long)b * 2048 + d) * 16 + n] = h[n];
}

// ---------------------------------------------------------------------------
extern "C" void kernel_launch(void* const* d_in, const int* in_sizes, int n_in,
                              void* d_out, int out_size, void* d_ws, size_t ws_size,
                              hipStream_t stream) {
    const float* x      = (const float*)d_in[0];   // [2,4096,1024]
    const float* hidden = (const float*)d_in[1];   // [2,2048,16]
    const float* W_in   = (const float*)d_in[2];   // [1024,4096]
    const float* conv_w = (const float*)d_in[3];   // [2048,1,4]
    const float* conv_b = (const float*)d_in[4];   // [2048]
    const float* W_x    = (const float*)d_in[5];   // [2048,96]
    const float* W_dt   = (const float*)d_in[6];   // [64,2048]
    const float* b_dt   = (const float*)d_in[7];   // [2048]
    const float* A_log  = (const float*)d_in[8];   // [2048,16]
    const float* Dvec   = (const float*)d_in[9];   // [2048]
    const float* W_out  = (const float*)d_in[10];  // [2048,1024]

    const int  MT = 2 * 4096;                      // B*L = 8192 rows
    const long A_PLANE = (long)MT * 2048;          // max A plane (elements)
    const long B_PLANE = (long)4096 * 1024;        // max Bt plane (elements)

    float* out    = (float*)d_out;                 // [2,4096,1024]
    float* fs_out = out + (long)MT * 1024;         // [2,2048,16]

    // workspace layout:
    float* ws     = (float*)d_ws;
    float* xz     = ws;                            // MT x 4096 f32
    float* x_conv = xz + (long)MT * 4096;          // MT x 2048 f32 (later y)
    float* x_dbl  = x_conv + (long)MT * 2048;      // MT x 96  f32
    bf16*  aHi    = (bf16*)(x_dbl + (long)MT * 96);
    bf16*  aLo    = aHi + A_PLANE;
    bf16*  bHi    = aLo + A_PLANE;
    bf16*  bLo    = bHi + B_PLANE;
    (void)ws_size; (void)in_sizes; (void)n_in; (void)out_size;

    dim3 blk(256);
    auto cdiv = [](long a, long b) { return (unsigned)((a + b - 1) / b); };

    // 1) xz = x @ W_in
    split_kernel <<<cdiv((long)MT * 1024, 256), blk, 0, stream>>>(x, aHi, aLo, (long)MT * 1024);
    splitT_kernel<<<cdiv((long)4096 * 1024, 256), blk, 0, stream>>>(W_in, bHi, bLo, 1024, 4096, (long)4096 * 1024);
    gemm_bf16x3_wmma<<<dim3(4096 / BN, MT / BM), blk, 0, stream>>>(
        aHi, aLo, 1024, bHi, bLo, 1024, xz, 4096, MT, 4096, 1024, nullptr, 0);

    // 2) depthwise conv + silu
    conv_silu_kernel<<<cdiv((long)MT * 2048, 256), blk, 0, stream>>>(xz, conv_w, conv_b, x_conv);

    // 3) x_dbl = x_conv @ W_x   (N=96, Npad=128)
    split_kernel <<<cdiv((long)MT * 2048, 256), blk, 0, stream>>>(x_conv, aHi, aLo, (long)MT * 2048);
    splitT_kernel<<<cdiv((long)128 * 2048, 256), blk, 0, stream>>>(W_x, bHi, bLo, 2048, 96, (long)128 * 2048);
    gemm_bf16x3_wmma<<<dim3(1, MT / BM), blk, 0, stream>>>(
        aHi, aLo, 2048, bHi, bLo, 2048, x_dbl, 96, MT, 96, 2048, nullptr, 0);

    // 4) dt = softplus(x_dbl[:, :64] @ W_dt + b_dt) -> dead xz cols (ldc 4096)
    split_kernel <<<cdiv((long)MT * 96, 256), blk, 0, stream>>>(x_dbl, aHi, aLo, (long)MT * 96);
    splitT_kernel<<<cdiv((long)2048 * 64, 256), blk, 0, stream>>>(W_dt, bHi, bLo, 64, 2048, (long)2048 * 64);
    gemm_bf16x3_wmma<<<dim3(2048 / BN, MT / BM), blk, 0, stream>>>(
        aHi, aLo, 96, bHi, bLo, 64, xz, 4096, MT, 2048, 64, b_dt, 1);

    // 5) selective scan (y gated by silu(z), in place over x_conv; fs -> tail)
    s6_scan_kernel<<<dim3(2048 / 256, 2), blk, 0, stream>>>(
        xz, x_conv, x_dbl, A_log, Dvec, hidden, fs_out);

    // 6) out = y @ W_out
    split_kernel <<<cdiv((long)MT * 2048, 256), blk, 0, stream>>>(x_conv, aHi, aLo, (long)MT * 2048);
    splitT_kernel<<<cdiv((long)1024 * 2048, 256), blk, 0, stream>>>(W_out, bHi, bLo, 2048, 1024, (long)1024 * 2048);
    gemm_bf16x3_wmma<<<dim3(1024 / BN, MT / BM), blk, 0, stream>>>(
        aHi, aLo, 2048, bHi, bLo, 2048, out, 1024, MT, 1024, 2048, nullptr, 0);
}
